// DeformConvWorldFeat_56848187129975
// MI455X (gfx1250) — matrix operand order repaired and checked by reading, not compiled, not run on hardware
//
#include <hip/hip_runtime.h>
#include <stdint.h>

// ---------------------------------------------------------------------------
// MI455X (gfx1250): all conv math as bf16 WMMA GEMMs (v_wmma_f32_16x16x32_bf16),
// im2col tiles staged in LDS. Dilated-conv / merge tiles are filled with
// global_load_async_to_lds_b128 issued tap-major and pipelined against the
// WMMA loop using staged s_wait_asynccnt + workgroup barriers. B-tile LDS
// addressing uses two hoisted base pointers so every ds_load_b128 in the
// k-loop is base+immediate (kills the WMMA->VALU WAR v_nop bubbles).
// ---------------------------------------------------------------------------

typedef __bf16 bf16_t;
typedef __attribute__((ext_vector_type(4)))  __bf16 v4bf;
typedef __attribute__((ext_vector_type(8)))  __bf16 v8bf;
typedef __attribute__((ext_vector_type(16))) __bf16 v16bf;
typedef __attribute__((ext_vector_type(8)))  float  v8f;

#define Hd    120
#define Wd    360
#define HW    43200        // Hd*Wd
#define Cc    128
#define NC    6
#define K9    1152         // Cc * 9
#define KP    1160         // LDS-padded K (row stride 2320B -> banks spread)
#define CM    768          // merge K = NC*Cc
#define CMP   776          // LDS-padded merge K (row stride 1552B)
#define HP    128          // Hd + 2*4 halo
#define WP    368          // Wd + 2*4 halo
#define NPIX  64           // pixels per GEMM tile
#define NTILE 675          // HW / NPIX
#define NPAIR 576          // NPIX * 9 taps
#define BT9_BYTES  (NPIX * KP * 2)   // 148480

static __device__ __forceinline__ v16bf cat8(v8bf a, v8bf b) {
  return __builtin_shufflevector(a, b, 0,1,2,3,4,5,6,7,8,9,10,11,12,13,14,15);
}
static __device__ __forceinline__ v8f zero8() {
  v8f z = {0.f,0.f,0.f,0.f,0.f,0.f,0.f,0.f};
  return z;
}
// CDNA5 async copy global->LDS, 16 bytes, tracked by ASYNCcnt. Low 32 bits of
// a generic LDS pointer are the LDS byte offset (shared-aperture layout).
static __device__ __forceinline__ void async_b128(void* lds, const void* g) {
  unsigned l = (unsigned)(uintptr_t)lds;
  unsigned long long a = (unsigned long long)(uintptr_t)g;
  asm volatile("global_load_async_to_lds_b128 %0, %1, off"
               :: "v"(l), "v"(a) : "memory");
}
template <int N>
static __device__ __forceinline__ void wait_async_le() {
#if defined(__has_builtin) && __has_builtin(__builtin_amdgcn_s_wait_asynccnt)
  __builtin_amdgcn_s_wait_asynccnt((short)N);
#else
  asm volatile("s_wait_asynccnt %0" :: "i"(N) : "memory");
#endif
}

// One 32-wide K step. A fragment from global (two 16B chunks, contiguous
// K = half*16 .. half*16+15). Four B fragments from LDS via the two
// hoisted base pointers bbA (columns nt=0/1) and bbB (nt=2/3); all eight
// ds_load_b128 use immediate offsets {0,16} / {16*LDK*2, 16*LDK*2+16}.
template <int LDK>
static __device__ __forceinline__ void gemm_ks(
    const bf16_t* __restrict__ A, const bf16_t* __restrict__ bbA,
    const bf16_t* __restrict__ bbB, int ks,
    v8f& a0, v8f& a1, v8f& a2, v8f& a3) {
  const bf16_t* ap = A + ks * 32;
  v16bf av = cat8(*(const v8bf*)ap, *(const v8bf*)(ap + 8));
  const bf16_t* pA = bbA + ks * 32;
  const bf16_t* pB = bbB + ks * 32;
  v16bf b0 = cat8(*(const v8bf*)(pA), *(const v8bf*)(pA + 8));
  a0 = __builtin_amdgcn_wmma_f32_16x16x32_bf16(false, av, false, b0, (short)0, a0, false, false);
  v16bf b1 = cat8(*(const v8bf*)(pA + 16 * LDK), *(const v8bf*)(pA + 16 * LDK + 8));
  a1 = __builtin_amdgcn_wmma_f32_16x16x32_bf16(false, av, false, b1, (short)0, a1, false, false);
  v16bf b2 = cat8(*(const v8bf*)(pB), *(const v8bf*)(pB + 8));
  a2 = __builtin_amdgcn_wmma_f32_16x16x32_bf16(false, av, false, b2, (short)0, a2, false, false);
  v16bf b3 = cat8(*(const v8bf*)(pB + 16 * LDK), *(const v8bf*)(pB + 16 * LDK + 8));
  a3 = __builtin_amdgcn_wmma_f32_16x16x32_bf16(false, av, false, b3, (short)0, a3, false, false);
}

// One pipeline phase: guarantee this wave's async loads for K-steps
// [KS0, KS0+4) have landed (in-order completion), barrier so the whole tile
// slice is valid, then run 4 K-steps of WMMA while later taps keep streaming.
template <int WAITN, int KS0, int LDK>
static __device__ __forceinline__ void async_phase(
    const bf16_t* __restrict__ A, const bf16_t* __restrict__ bbA,
    const bf16_t* __restrict__ bbB,
    v8f& a0, v8f& a1, v8f& a2, v8f& a3) {
  wait_async_le<WAITN>();
  __syncthreads();
  #pragma unroll
  for (int j = 0; j < 4; ++j)
    gemm_ks<LDK>(A, bbA, bbB, KS0 + j, a0, a1, a2, a3);
}

// ---------------------------------------------------------------------------
// f = x[cam] + pos_emb  ->  channel-last bf16 [cam][pix][128] via LDS transpose
__global__ __launch_bounds__(256) void prep_feat(
    const float* __restrict__ x, const float* __restrict__ pos,
    bf16_t* __restrict__ featN) {
  __shared__ float tile[NPIX][Cc + 1];
  const int cam  = blockIdx.y;
  const int pix0 = blockIdx.x * NPIX;
  const int t    = threadIdx.x;
  const float* xc = x + (size_t)cam * Cc * HW;
  #pragma unroll 4
  for (int i = 0; i < 32; ++i) {          // coalesced planar reads
    int c = i * 4 + (t >> 6);
    int p = t & 63;
    tile[p][c] = xc[(size_t)c * HW + pix0 + p] + pos[(size_t)c * HW + pix0 + p];
  }
  __syncthreads();
  #pragma unroll
  for (int j = 0; j < 8; ++j) {           // contiguous channel-last writes
    int p  = j * 8 + (t >> 5);
    int c0 = (t & 31) * 4;
    v4bf v;
    v[0] = (bf16_t)tile[p][c0 + 0]; v[1] = (bf16_t)tile[p][c0 + 1];
    v[2] = (bf16_t)tile[p][c0 + 2]; v[3] = (bf16_t)tile[p][c0 + 3];
    *(v4bf*)(featN + ((size_t)cam * HW + pix0 + p) * Cc + c0) = v;
  }
}

// ---------------------------------------------------------------------------
// off[cam][pix][18] = w_off[cam] (18x128) . f(pix) + b_off[cam]
__global__ __launch_bounds__(256) void offsets_kernel(
    const bf16_t* __restrict__ featN, const float* __restrict__ w_off,
    const float* __restrict__ b_off, float* __restrict__ off) {
  __shared__ float wl[18 * Cc];
  const int cam = blockIdx.y;
  for (int i = threadIdx.x; i < 18 * Cc; i += 256)
    wl[i] = w_off[cam * 18 * Cc + i];
  __syncthreads();
  int pix = blockIdx.x * 256 + threadIdx.x;
  if (pix >= HW) return;
  const bf16_t* f = featN + ((size_t)cam * HW + pix) * Cc;
  float acc[18];
  #pragma unroll
  for (int j = 0; j < 18; ++j) acc[j] = b_off[cam * 18 + j];
  for (int c = 0; c < Cc; ++c) {
    float v = (float)f[c];
    #pragma unroll
    for (int j = 0; j < 18; ++j) acc[j] += wl[j * Cc + c] * v;   // LDS broadcast
  }
  float* o = off + ((size_t)cam * HW + pix) * 18;
  #pragma unroll
  for (int j = 0; j < 18; ++j) o[j] = acc[j];
}

// ---------------------------------------------------------------------------
// [O][C][3][3] fp32 -> bf16 A-matrix [O][K], K = k*128 + c (k = kh*3+kw)
__global__ void reorder_w(const float* __restrict__ w, bf16_t* __restrict__ wA,
                          int total) {
  int i = blockIdx.x * 256 + threadIdx.x;
  if (i >= total) return;
  int o = i / K9, r = i - o * K9, k = r >> 7, c = r & 127;
  wA[i] = (bf16_t)w[((size_t)o * Cc + c) * 9 + k];
}
__global__ void cvt_bf16(const float* __restrict__ in, bf16_t* __restrict__ out,
                         int n) {
  int i = blockIdx.x * 256 + threadIdx.x;
  if (i < n) out[i] = (bf16_t)in[i];
}
__global__ void fill_zero_u32(uint32_t* __restrict__ p, long n) {
  long i = (long)blockIdx.x * blockDim.x + threadIdx.x;
  long s = (long)gridDim.x * blockDim.x;
  for (; i < n; i += s) p[i] = 0u;
}

// ---------------------------------------------------------------------------
// Deformable conv as WMMA GEMM. Stage 0 computes per-(pixel,tap) clamped
// corner indices + validity-folded bilinear weights once into LDS. Stage 1
// builds the bf16 im2col tile with branchless vector loads. Stage 2 is the
// 128x64x1152 WMMA GEMM.
__global__ __launch_bounds__(256) void deform_gemm(
    const bf16_t* __restrict__ featN, const float* __restrict__ off,
    const bf16_t* __restrict__ wAdef, const float* __restrict__ b_def,
    bf16_t* __restrict__ dcat) {
  extern __shared__ __align__(16) char smem[];
  bf16_t* Bt   = (bf16_t*)smem;                          // [NPIX][KP]
  int4*   midx = (int4*)(smem + BT9_BYTES);              // [NPAIR] corner idx
  float4* mwgt = (float4*)(smem + BT9_BYTES + NPAIR*16); // [NPAIR] weights
  const int cam  = blockIdx.y;
  const int pix0 = blockIdx.x * NPIX;
  const bf16_t* feat = featN + (size_t)cam * HW * Cc;
  const float*  offc = off  + (size_t)cam * HW * 18;
  const int t = threadIdx.x;
  // ---- stage 0: bilinear metadata, one thread per (pixel,tap) pair
  for (int pair = t; pair < NPAIR; pair += 256) {
    int p = pair / 9, k = pair - p * 9;
    int pix = pix0 + p;
    int y = pix / Wd, xx = pix - y * Wd;
    int kh = k / 3, kw = k - kh * 3;
    float py = offc[(size_t)pix * 18 + k * 2]     + (float)(y  - 1 + kh);
    float px = offc[(size_t)pix * 18 + k * 2 + 1] + (float)(xx - 1 + kw);
    float fy = floorf(py), fx = floorf(px);
    int y0 = (int)fy, x0 = (int)fx;
    float ty = py - fy, tx = px - fx;
    int   id[4]; float wg[4];
    float tw[4] = {(1.f-ty)*(1.f-tx), (1.f-ty)*tx, ty*(1.f-tx), ty*tx};
    #pragma unroll
    for (int q = 0; q < 4; ++q) {
      int yy = y0 + (q >> 1), xq = x0 + (q & 1);
      bool ok = (yy >= 0) & (yy < Hd) & (xq >= 0) & (xq < Wd);
      int yc = min(max(yy, 0), Hd - 1), xc = min(max(xq, 0), Wd - 1);
      id[q] = yc * Wd + xc;
      wg[q] = ok ? tw[q] : 0.f;
    }
    midx[pair] = make_int4(id[0], id[1], id[2], id[3]);
    mwgt[pair] = make_float4(wg[0], wg[1], wg[2], wg[3]);
  }
  __syncthreads();
  // ---- stage 1: blend 8 channels per thread per pair (branchless)
  const int g = t >> 4, l16 = t & 15, c0 = l16 * 8;
  for (int it = 0; it < 36; ++it) {
    int pair = it * 16 + g;
    int p = pair / 9, k = pair - p * 9;
    int4   id = midx[pair];          // broadcast LDS reads across 16 threads
    float4 wq = mwgt[pair];
    int   idx[4] = {id.x, id.y, id.z, id.w};
    float ww[4]  = {wq.x, wq.y, wq.z, wq.w};
    float a[8] = {0.f,0.f,0.f,0.f,0.f,0.f,0.f,0.f};
    #pragma unroll
    for (int q = 0; q < 4; ++q) {
      v8bf vv = *(const v8bf*)(feat + (size_t)idx[q] * Cc + c0);
      #pragma unroll
      for (int i = 0; i < 8; ++i) a[i] += ww[q] * (float)vv[i];
    }
    v8bf st;
    #pragma unroll
    for (int i = 0; i < 8; ++i) st[i] = (bf16_t)a[i];
    *(v8bf*)(Bt + p * KP + k * Cc + c0) = st;
  }
  __syncthreads();
  // ---- stage 2: GEMM, wave wv owns rows [wv*16, wv*16+16) x 64 columns
  const int wv = t >> 5, l = t & 31;
  const int r16 = l & 15, half = l >> 4;
  const bf16_t* A = wAdef + (size_t)cam * Cc * K9
                  + (size_t)(wv * 16 + r16) * K9 + half * 16;
  const bf16_t* bbA = Bt + r16 * KP + half * 16;   // columns nt=0/1
  const bf16_t* bbB = bbA + 32 * KP;               // columns nt=2/3
  v8f acc0 = zero8(), acc1 = zero8(), acc2 = zero8(), acc3 = zero8();
  for (int ks = 0; ks < 36; ++ks)
    gemm_ks<KP>(A, bbA, bbB, ks, acc0, acc1, acc2, acc3);
  // ---- store: rows r are consecutive channels -> one 16B store per frag
  v8f accs[4] = {acc0, acc1, acc2, acc3};
  const int obase = wv * 16 + half * 8;
  #pragma unroll
  for (int nt = 0; nt < 4; ++nt) {
    int pix = pix0 + nt * 16 + r16;
    v8bf ov;
    #pragma unroll
    for (int r = 0; r < 8; ++r)
      ov[r] = (bf16_t)fmaxf(accs[nt][r] + b_def[cam * Cc + obase + r], 0.f);
    *(v8bf*)(dcat + (size_t)pix * CM + cam * Cc + obase) = ov;
  }
}

// ---------------------------------------------------------------------------
// Merge 1x1 conv: K=768 GEMM. B tile streamed with async-to-LDS in K-major
// order and pipelined: 6 phases x (staged asynccnt wait + 4 WMMA K-steps).
__global__ __launch_bounds__(256) void merge_gemm(
    const bf16_t* __restrict__ dcat, const bf16_t* __restrict__ wAm,
    const float* __restrict__ b_merge, bf16_t* __restrict__ outPad) {
  extern __shared__ __align__(16) char smem[];
  bf16_t* Bt = (bf16_t*)smem;                 // [NPIX][CMP]
  const int pix0 = blockIdx.x * NPIX;
  const int t = threadIdx.x;
  // K-major issue: per thread, each iteration covers one 32-wide K-step,
  // so asynccnt <= 24-4*(phase+1) implies that phase's K slice has landed.
  for (int it = 0; it < 24; ++it) {
    int chunk = it * 256 + t;
    int cblk = chunk >> 6, p = chunk & 63;
    int c0 = cblk * 8;
    async_b128(Bt + p * CMP + c0, dcat + (size_t)(pix0 + p) * CM + c0);
  }
  const int wv = t >> 5, l = t & 31;
  const int r16 = l & 15, half = l >> 4;
  const bf16_t* A = wAm + (size_t)(wv * 16 + r16) * CM + half * 16;
  const bf16_t* bbA = Bt + r16 * CMP + half * 16;
  const bf16_t* bbB = bbA + 32 * CMP;
  v8f acc0 = zero8(), acc1 = zero8(), acc2 = zero8(), acc3 = zero8();
  async_phase<20,  0, CMP>(A, bbA, bbB, acc0, acc1, acc2, acc3);
  async_phase<16,  4, CMP>(A, bbA, bbB, acc0, acc1, acc2, acc3);
  async_phase<12,  8, CMP>(A, bbA, bbB, acc0, acc1, acc2, acc3);
  async_phase< 8, 12, CMP>(A, bbA, bbB, acc0, acc1, acc2, acc3);
  async_phase< 4, 16, CMP>(A, bbA, bbB, acc0, acc1, acc2, acc3);
  async_phase< 0, 20, CMP>(A, bbA, bbB, acc0, acc1, acc2, acc3);
  v8f accs[4] = {acc0, acc1, acc2, acc3};
  const int obase = wv * 16 + half * 8;
  #pragma unroll
  for (int nt = 0; nt < 4; ++nt) {
    int pix = pix0 + nt * 16 + r16;
    int y = pix / Wd, x = pix - y * Wd;
    size_t pidx = (size_t)(y + 4) * WP + (x + 4);
    v8bf ov;
    #pragma unroll
    for (int r = 0; r < 8; ++r)
      ov[r] = (bf16_t)fmaxf(accs[nt][r] + b_merge[obase + r], 0.f);
    *(v8bf*)(outPad + pidx * Cc + obase) = ov;
  }
}

// ---------------------------------------------------------------------------
// 3x3 dilated conv (dil = 1/2/4): zero-haloed input makes every tap a pure
// copy, issued tap-major as async b128 and pipelined against the WMMA loop
// (9 phases). Last pass writes the fp32 planar output.
__global__ __launch_bounds__(256) void dilconv_gemm(
    const bf16_t* __restrict__ inPad, const bf16_t* __restrict__ wA,
    const float* __restrict__ bias, int dil,
    bf16_t* __restrict__ outPad, float* __restrict__ outF32) {
  extern __shared__ __align__(16) char smem[];
  bf16_t* Bt = (bf16_t*)smem;                 // [NPIX][KP]
  const int pix0 = blockIdx.x * NPIX;
  const int t = threadIdx.x;
  const int g = t >> 4, l16 = t & 15;
  // tap-major issue: pair = k*64 + p -> per thread, 4 loads per tap, in order
  for (int it = 0; it < 36; ++it) {
    int pair = it * 16 + g;
    int k = pair >> 6, p = pair & 63;
    int pix = pix0 + p;
    int y = pix / Wd, x = pix - y * Wd;
    int kh = k / 3, kw = k - kh * 3;
    int iy = y + 4 + (kh - 1) * dil;
    int ix = x + 4 + (kw - 1) * dil;
    async_b128(Bt + p * KP + k * Cc + l16 * 8,
               inPad + ((size_t)iy * WP + ix) * Cc + l16 * 8);
  }
  const int wv = t >> 5, l = t & 31;
  const int r16 = l & 15, half = l >> 4;
  const bf16_t* A = wA + (size_t)(wv * 16 + r16) * K9 + half * 16;
  const bf16_t* bbA = Bt + r16 * KP + half * 16;
  const bf16_t* bbB = bbA + 32 * KP;
  v8f acc0 = zero8(), acc1 = zero8(), acc2 = zero8(), acc3 = zero8();
  async_phase<32,  0, KP>(A, bbA, bbB, acc0, acc1, acc2, acc3);
  async_phase<28,  4, KP>(A, bbA, bbB, acc0, acc1, acc2, acc3);
  async_phase<24,  8, KP>(A, bbA, bbB, acc0, acc1, acc2, acc3);
  async_phase<20, 12, KP>(A, bbA, bbB, acc0, acc1, acc2, acc3);
  async_phase<16, 16, KP>(A, bbA, bbB, acc0, acc1, acc2, acc3);
  async_phase<12, 20, KP>(A, bbA, bbB, acc0, acc1, acc2, acc3);
  async_phase< 8, 24, KP>(A, bbA, bbB, acc0, acc1, acc2, acc3);
  async_phase< 4, 28, KP>(A, bbA, bbB, acc0, acc1, acc2, acc3);
  async_phase< 0, 32, KP>(A, bbA, bbB, acc0, acc1, acc2, acc3);
  v8f accs[4] = {acc0, acc1, acc2, acc3};
  const int obase = wv * 16 + half * 8;
  #pragma unroll
  for (int nt = 0; nt < 4; ++nt) {
    int pix = pix0 + nt * 16 + r16;
    if (outF32) {
      #pragma unroll
      for (int r = 0; r < 8; ++r)
        outF32[(size_t)(obase + r) * HW + pix] =
            fmaxf(accs[nt][r] + bias[obase + r], 0.f);
    } else {
      int y = pix / Wd, x = pix - y * Wd;
      v8bf ov;
      #pragma unroll
      for (int r = 0; r < 8; ++r)
        ov[r] = (bf16_t)fmaxf(accs[nt][r] + bias[obase + r], 0.f);
      *(v8bf*)(outPad + ((size_t)(y + 4) * WP + (x + 4)) * Cc + obase) = ov;
    }
  }
}

// ---------------------------------------------------------------------------
extern "C" void kernel_launch(void* const* d_in, const int* in_sizes, int n_in,
                              void* d_out, int out_size, void* d_ws, size_t ws_size,
                              hipStream_t stream) {
  (void)in_sizes; (void)n_in; (void)out_size; (void)ws_size;
  const float* x       = (const float*)d_in[0];
  const float* pos     = (const float*)d_in[1];
  const float* w_off   = (const float*)d_in[2];
  const float* b_off   = (const float*)d_in[3];
  const float* w_def   = (const float*)d_in[4];
  const float* b_def   = (const float*)d_in[5];
  const float* w_merge = (const float*)d_in[6];
  const float* b_merge = (const float*)d_in[7];
  const float* w1 = (const float*)d_in[8];  const float* b1 = (const float*)d_in[9];
  const float* w2 = (const float*)d_in[10]; const float* b2 = (const float*)d_in[11];
  const float* w3 = (const float*)d_in[12]; const float* b3 = (const float*)d_in[13];
  float* out = (float*)d_out;

  // --- workspace carve-out (~190 MB, 256B aligned slices) ---
  char* ws = (char*)d_ws;
  auto carve = [&](size_t bytes) -> char* {
    char* p = ws;
    ws += (bytes + 255) & ~(size_t)255;
    return p;
  };
  bf16_t* featN = (bf16_t*)carve((size_t)NC * HW * Cc * 2);   // 66.4 MB
  float*  off   = (float*) carve((size_t)NC * HW * 18 * 4);   // 18.7 MB
  bf16_t* wAdef = (bf16_t*)carve((size_t)NC * Cc * K9 * 2);   // 1.8 MB
  bf16_t* wA1   = (bf16_t*)carve((size_t)Cc * K9 * 2);
  bf16_t* wA2   = (bf16_t*)carve((size_t)Cc * K9 * 2);
  bf16_t* wA3   = (bf16_t*)carve((size_t)Cc * K9 * 2);
  bf16_t* wAm   = (bf16_t*)carve((size_t)Cc * CM * 2);
  bf16_t* dcat  = (bf16_t*)carve((size_t)HW * CM * 2);        // 66.4 MB
  const size_t padElems = (size_t)HP * WP * Cc;               // haloed bf16 maps
  bf16_t* pad0  = (bf16_t*)carve(padElems * 2);               // merge out
  bf16_t* pad1  = (bf16_t*)carve(padElems * 2);               // conv1 out
  bf16_t* pad2  = (bf16_t*)carve(padElems * 2);               // conv2 out

  // zero haloed buffers (interior rewritten each call; halo must be 0)
  fill_zero_u32<<<2048, 256, 0, stream>>>((uint32_t*)pad0, (long)(padElems / 2));
  fill_zero_u32<<<2048, 256, 0, stream>>>((uint32_t*)pad1, (long)(padElems / 2));
  fill_zero_u32<<<2048, 256, 0, stream>>>((uint32_t*)pad2, (long)(padElems / 2));

  // weight reorder / convert
  {
    int tdef = NC * Cc * K9;                  // 6*128*1152
    reorder_w<<<(tdef + 255) / 256, 256, 0, stream>>>(w_def, wAdef, tdef);
    int tc = Cc * K9;
    reorder_w<<<(tc + 255) / 256, 256, 0, stream>>>(w1, wA1, tc);
    reorder_w<<<(tc + 255) / 256, 256, 0, stream>>>(w2, wA2, tc);
    reorder_w<<<(tc + 255) / 256, 256, 0, stream>>>(w3, wA3, tc);
    int tm = Cc * CM;
    cvt_bf16<<<(tm + 255) / 256, 256, 0, stream>>>(w_merge, wAm, tm);
  }

  // feature prep + offsets
  prep_feat<<<dim3(NTILE, NC), 256, 0, stream>>>(x, pos, featN);
  offsets_kernel<<<dim3((HW + 255) / 256, NC), 256, 0, stream>>>(featN, w_off, b_off, off);

  // deformable conv GEMMs (bilinear im2col in LDS -> WMMA)
  const size_t ldsDef   = (size_t)BT9_BYTES + (size_t)NPAIR * 32; // +meta
  const size_t ldsConv  = (size_t)BT9_BYTES;
  const size_t ldsMerge = (size_t)NPIX * CMP * 2;
  deform_gemm<<<dim3(NTILE, NC), 256, ldsDef, stream>>>(featN, off, wAdef, b_def, dcat);

  // merge + dilated conv stack (async-to-LDS pipelined GEMMs)
  merge_gemm<<<NTILE, 256, ldsMerge, stream>>>(dcat, wAm, b_merge, pad0);
  dilconv_gemm<<<NTILE, 256, ldsConv, stream>>>(pad0, wA1, b1, 1, pad1, nullptr);
  dilconv_gemm<<<NTILE, 256, ldsConv, stream>>>(pad1, wA2, b2, 2, pad2, nullptr);
  dilconv_gemm<<<NTILE, 256, ldsConv, stream>>>(pad2, wA3, b3, 4, nullptr, out);
}
